// TFFastSpeechSelfAttention_11871289606216
// MI455X (gfx1250) — matrix operand order repaired and checked by reading, not compiled, not run on hardware
//
#include <hip/hip_runtime.h>
#include <hip/hip_bf16.h>

// ---------------------------------------------------------------------------
// Types
// ---------------------------------------------------------------------------
typedef __bf16 bf16_t;
typedef __attribute__((ext_vector_type(8)))  __bf16 v8bf;
typedef __attribute__((ext_vector_type(16))) __bf16 v16bf;
typedef __attribute__((ext_vector_type(8)))  float  v8f;
typedef int v4i32 __attribute__((vector_size(16)));   // matches builtin param type

__device__ __forceinline__ bf16_t f2bf(float f) {
    // round-to-nearest-even fp32 -> bf16 without relying on __bf16 arithmetic
    unsigned x = __float_as_uint(f);
    unsigned r = x + 0x7FFFu + ((x >> 16) & 1u);
    unsigned short u = (unsigned short)(r >> 16);
    return *reinterpret_cast<bf16_t*>(&u);
}

union FragBF { v16bf v; v8bf h[2]; };

// ---------------------------------------------------------------------------
// CDNA5 async global->LDS copy path (ASYNCcnt-tracked), with fallback.
// Builtin signature (from hipcc diagnostic): param0 = AS(1) v4i32*, so pass
// 16-byte int-vector pointers in the right address spaces.
// ---------------------------------------------------------------------------
#if defined(__has_builtin)
#if __has_builtin(__builtin_amdgcn_global_load_async_to_lds_b128) && \
    __has_builtin(__builtin_amdgcn_s_wait_asynccnt)
#define USE_ASYNC_LDS 1
#endif
#endif

#if defined(USE_ASYNC_LDS)
__device__ __forceinline__ void async_copy_b128(const void* g, void* l) {
    __builtin_amdgcn_global_load_async_to_lds_b128(
        (__attribute__((address_space(1))) v4i32*)g,
        (__attribute__((address_space(3))) v4i32*)l, 0, 0);
}
#define WAIT_ASYNC_3() __builtin_amdgcn_s_wait_asynccnt(3)
#define WAIT_ASYNC_0() __builtin_amdgcn_s_wait_asynccnt(0)
#else
#define WAIT_ASYNC_3()
#define WAIT_ASYNC_0()
#endif

// ---------------------------------------------------------------------------
// fp32 -> bf16 conversion (grid-stride)
// ---------------------------------------------------------------------------
__global__ void cvt_f32_bf16(const float* __restrict__ s, bf16_t* __restrict__ d, int n) {
    int i = blockIdx.x * blockDim.x + threadIdx.x;
    int stride = gridDim.x * blockDim.x;
    for (; i < n; i += stride) d[i] = f2bf(s[i]);
}

// ---------------------------------------------------------------------------
// 128x64-tile bf16 WMMA GEMM, K = 1024, K-step 32, 256 threads (8 wave32).
// Wave w: rows (w&3)*32 .. +31, cols (w>>2)*32 .. +31 -> four 16x16 WMMA tiles.
// Double-buffered LDS; tile fetches use GLOBAL_LOAD_ASYNC_TO_LDS_B128 when
// available so the next stage streams in while the current stage's WMMAs run.
//
// MODE 0: C(bf16, MxN row-major) = A@B + bias          (Q, V projections)
// MODE 1: C(bf16) = A@B + bias, stored transposed per batch: C[b][n][s]  (K^T)
// MODE 2: C(f32)  = (A@B)*0.125 + (1-mask)*-1e9, per-batch (blockIdx.z)  (QK^T)
// MODE 3: C(f32)  = A@B written permuted to (B,H,Sq,DH) flat output      (P@V)
// ---------------------------------------------------------------------------
template<int MODE>
__global__ __launch_bounds__(256) void gemm_wmma_128x64(
    const bf16_t* __restrict__ A, const bf16_t* __restrict__ Bm,
    const float* __restrict__ aux, void* __restrict__ Cout)
{
    constexpr int SA = 40;   // LDS row stride (bf16) for A tile 128x32 -> 80B rows
    constexpr int SB = 72;   // LDS row stride (bf16) for B tile  32x64 -> 144B rows
    __shared__ alignas(16) bf16_t sA[2][128 * SA];
    __shared__ alignas(16) bf16_t sB[2][32 * SB];

    const int t    = threadIdx.x;
    const int wave = t >> 5, lane = t & 31;
    const int mstrip = wave & 3;   // 32-row strip within 128-row block tile
    const int nh     = wave >> 2;  // 32-col half  within  64-col block tile
    const int z  = blockIdx.z;
    const int gm0 = blockIdx.y * 128;
    const int gn0 = blockIdx.x * 64;

    const bf16_t* Ab = A;
    const bf16_t* Bb = Bm;
    const float*  Xb = aux;
    if (MODE >= 2) {
        Ab += (size_t)z << 20;                 // per-batch 1024x1024 slabs
        Bb += (size_t)z << 20;
        if (MODE == 2) Xb += (size_t)z << 20;  // attention mask slab
    }

    // Cooperative tile mapping (per stage: 3 x b128 per thread)
    const int a_row = t >> 1, a_c = (t & 1) * 16;   // A: 128 rows x 32 cols
    const int b_row = t >> 3, b_c = (t & 7) * 8;    // B:  32 rows x 64 cols

    v8f acc[2][2] = {};

    auto issue = [&](int buf, int k0) {
        const bf16_t* ga = Ab + (size_t)(gm0 + a_row) * 1024 + k0 + a_c;
        bf16_t*       la = &sA[0][0] + (size_t)buf * (128 * SA) + a_row * SA + a_c;
        const bf16_t* gb = Bb + (size_t)(k0 + b_row) * 1024 + gn0 + b_c;
        bf16_t*       lb = &sB[0][0] + (size_t)buf * (32 * SB) + b_row * SB + b_c;
#if defined(USE_ASYNC_LDS)
        async_copy_b128(ga,     la);
        async_copy_b128(ga + 8, la + 8);
        async_copy_b128(gb,     lb);
#else
        uint4 av0 = *(const uint4*)(ga);
        uint4 av1 = *(const uint4*)(ga + 8);
        uint4 bv  = *(const uint4*)(gb);
        *(uint4*)(la)     = av0;
        *(uint4*)(la + 8) = av1;
        *(uint4*)(lb)     = bv;
#endif
    };

    auto compute = [&](const bf16_t* __restrict__ cA, const bf16_t* __restrict__ cB) {
        // A fragments (16x32): lane m=L&15; half h=L>>4 holds K h*8..h*8+7, 16+h*8..
        const int m = lane & 15, hh = lane >> 4;
        const bf16_t* ar0 = cA + (mstrip * 32 + m) * SA;
        const bf16_t* ar1 = ar0 + 16 * SA;
        FragBF a0, a1;
        a0.h[0] = *(const v8bf*)(ar0 + hh * 8);
        a0.h[1] = *(const v8bf*)(ar0 + 16 + hh * 8);
        a1.h[0] = *(const v8bf*)(ar1 + hh * 8);
        a1.h[1] = *(const v8bf*)(ar1 + 16 + hh * 8);
        // B fragments (32x16): lane L holds column N=L&15, K in same order as A
        const bf16_t* br = cB + nh * 32 + m;
        FragBF b0, b1;
#pragma unroll
        for (int s = 0; s < 16; ++s) {
            const int kk = (s < 8) ? (hh * 8 + s) : (16 + hh * 8 + (s - 8));
            b0.v[s] = br[kk * SB];
            b1.v[s] = br[kk * SB + 16];
        }

        acc[0][0] = __builtin_amdgcn_wmma_f32_16x16x32_bf16(false, a0.v, false, b0.v,
                                                            (short)0, acc[0][0], false, false);
        acc[0][1] = __builtin_amdgcn_wmma_f32_16x16x32_bf16(false, a0.v, false, b1.v,
                                                            (short)0, acc[0][1], false, false);
        acc[1][0] = __builtin_amdgcn_wmma_f32_16x16x32_bf16(false, a1.v, false, b0.v,
                                                            (short)0, acc[1][0], false, false);
        acc[1][1] = __builtin_amdgcn_wmma_f32_16x16x32_bf16(false, a1.v, false, b1.v,
                                                            (short)0, acc[1][1], false, false);
    };

    // Double-buffered main loop over K (32 steps of 32), unrolled by 2 so the
    // ping-pong buffer index is compile-time constant.
    issue(0, 0);
    for (int ks = 0; ks < 32; ks += 2) {
        issue(1, (ks + 1) * 32);      // ks+1 <= 31 always
        WAIT_ASYNC_3();               // stage ks (buf0) complete (in-order)
        __syncthreads();
        compute(sA[0], sB[0]);
        __syncthreads();              // buf0 reads done before it is refilled

        if (ks + 2 < 32) {
            issue(0, (ks + 2) * 32);
            WAIT_ASYNC_3();
        } else {
            WAIT_ASYNC_0();
        }
        __syncthreads();
        compute(sA[1], sB[1]);
        __syncthreads();
    }

    // C/D layout: VGPR r -> M=r (lanes 0-15) / M=r+8 (lanes 16-31); N = lane&15
    const int col   = lane & 15;
    const int rbase = (lane < 16) ? 0 : 8;
#pragma unroll
    for (int sub = 0; sub < 2; ++sub) {
#pragma unroll
        for (int r = 0; r < 8; ++r) {
            const int gm = gm0 + mstrip * 32 + sub * 16 + rbase + r;
#pragma unroll
            for (int tt = 0; tt < 2; ++tt) {
                const int gn  = gn0 + nh * 32 + tt * 16 + col;
                const float v = acc[sub][tt][r];
                if (MODE == 0) {
                    ((bf16_t*)Cout)[(size_t)gm * 1024 + gn] = f2bf(v + aux[gn]);
                } else if (MODE == 1) {
                    const int b = gm >> 10, s = gm & 1023;
                    ((bf16_t*)Cout)[((size_t)b * 1024 + gn) * 1024 + s] = f2bf(v + aux[gn]);
                } else if (MODE == 2) {
                    float* C = (float*)Cout + ((size_t)z << 20);
                    C[(size_t)gm * 1024 + gn] =
                        v * 0.125f + (1.0f - Xb[(size_t)gm * 1024 + gn]) * -1.0e9f;
                } else {  // MODE 3: out[((z*16 + h)*1024 + q)*64 + d]
                    const int h = gn >> 6, d = gn & 63;
                    ((float*)Cout)[(((size_t)z * 16 + h) * 1024 + gm) * 64 + d] = v;
                }
            }
        }
    }
}

// ---------------------------------------------------------------------------
// Row softmax over 1024 columns; one 256-thread block per row; bf16 output.
// ---------------------------------------------------------------------------
__global__ __launch_bounds__(256) void softmax_rows(const float* __restrict__ scores,
                                                    bf16_t* __restrict__ probs)
{
    const size_t row = blockIdx.x;
    const float* s = scores + row * 1024;
    const int t = threadIdx.x;
    __shared__ float red[8];

    float vals[4];
    float mx = -3.4e38f;
#pragma unroll
    for (int i = 0; i < 4; ++i) {
        vals[i] = s[t + i * 256];
        mx = fmaxf(mx, vals[i]);
    }
#pragma unroll
    for (int off = 16; off > 0; off >>= 1) mx = fmaxf(mx, __shfl_xor(mx, off, 32));
    if ((t & 31) == 0) red[t >> 5] = mx;
    __syncthreads();
#pragma unroll
    for (int j = 0; j < 8; ++j) mx = fmaxf(mx, red[j]);
    __syncthreads();  // done reading red before reuse

    float e[4], sum = 0.0f;
#pragma unroll
    for (int i = 0; i < 4; ++i) { e[i] = __expf(vals[i] - mx); sum += e[i]; }
#pragma unroll
    for (int off = 16; off > 0; off >>= 1) sum += __shfl_xor(sum, off, 32);
    if ((t & 31) == 0) red[t >> 5] = sum;
    __syncthreads();
    float tot = 0.0f;
#pragma unroll
    for (int j = 0; j < 8; ++j) tot += red[j];
    const float inv = 1.0f / tot;
#pragma unroll
    for (int i = 0; i < 4; ++i) probs[row * 1024 + t + i * 256] = f2bf(e[i] * inv);
}

// ---------------------------------------------------------------------------
// Launch
// ---------------------------------------------------------------------------
extern "C" void kernel_launch(void* const* d_in, const int* in_sizes, int n_in,
                              void* d_out, int out_size, void* d_ws, size_t ws_size,
                              hipStream_t stream)
{
    (void)in_sizes; (void)n_in; (void)out_size; (void)ws_size;

    const float* hs  = (const float*)d_in[0];   // (8,1024,1024)
    const float* rhs = (const float*)d_in[1];   // (8,1024,1024)
    const float* msk = (const float*)d_in[2];   // (8,1024,1024)
    const float* Wq  = (const float*)d_in[3];
    const float* bq  = (const float*)d_in[4];
    const float* Wk  = (const float*)d_in[5];
    const float* bk  = (const float*)d_in[6];
    const float* Wv  = (const float*)d_in[7];
    const float* bv  = (const float*)d_in[8];
    float* out = (float*)d_out;                 // (8,1024,1024) flat (B,H,Sq,DH)

    char* ws = (char*)d_ws;
    size_t off = 0;
    auto alloc = [&](size_t bytes) -> char* {
        char* p = ws + off;
        off += (bytes + 255) & ~(size_t)255;
        return p;
    };
    const size_t NBIG = 8ull * 1024 * 1024;           // 8M elements
    bf16_t* hsb    = (bf16_t*)alloc(NBIG * 2);        // hs bf16
    bf16_t* rhsb   = (bf16_t*)alloc(NBIG * 2);        // rhs bf16
    bf16_t* wqb    = (bf16_t*)alloc(1024 * 1024 * 2);
    bf16_t* wkb    = (bf16_t*)alloc(1024 * 1024 * 2);
    bf16_t* wvb    = (bf16_t*)alloc(1024 * 1024 * 2);
    bf16_t* Qb     = (bf16_t*)alloc(NBIG * 2);        // (8192,1024) bf16
    bf16_t* Ktb    = (bf16_t*)alloc(NBIG * 2);        // (8,1024d,1024s) bf16
    bf16_t* Vb     = (bf16_t*)alloc(NBIG * 2);        // (8192,1024) bf16
    float*  scores = (float*)alloc(NBIG * 4);         // (8,1024,1024) f32
    bf16_t* probs  = (bf16_t*)alloc(NBIG * 2);        // (8,1024,1024) bf16

    // 1) fp32 -> bf16 conversions
    cvt_f32_bf16<<<2048, 256, 0, stream>>>(hs,  hsb,  (int)NBIG);
    cvt_f32_bf16<<<2048, 256, 0, stream>>>(rhs, rhsb, (int)NBIG);
    cvt_f32_bf16<<<1024, 256, 0, stream>>>(Wq, wqb, 1024 * 1024);
    cvt_f32_bf16<<<1024, 256, 0, stream>>>(Wk, wkb, 1024 * 1024);
    cvt_f32_bf16<<<1024, 256, 0, stream>>>(Wv, wvb, 1024 * 1024);

    // 2) projections (WMMA): Q, K^T, V   (M=8192 -> 64 row-blocks of 128)
    gemm_wmma_128x64<0><<<dim3(16, 64, 1), 256, 0, stream>>>(hsb,  wqb, bq, (void*)Qb);
    gemm_wmma_128x64<1><<<dim3(16, 64, 1), 256, 0, stream>>>(rhsb, wkb, bk, (void*)Ktb);
    gemm_wmma_128x64<0><<<dim3(16, 64, 1), 256, 0, stream>>>(rhsb, wvb, bv, (void*)Vb);

    // 3) head-summed scores = Q@K^T * 0.125 + mask extension (per batch)
    gemm_wmma_128x64<2><<<dim3(16, 8, 8), 256, 0, stream>>>(Qb, Ktb, msk, (void*)scores);

    // 4) softmax rows -> bf16 probs
    softmax_rows<<<8192, 256, 0, stream>>>(scores, probs);

    // 5) ctx = P@V, written permuted to (B,H,Sq,DH) flat output
    gemm_wmma_128x64<3><<<dim3(16, 8, 8), 256, 0, stream>>>(probs, Vb, bq /*unused*/, (void*)out);
}
